// Summarizer_59785944760779
// MI455X (gfx1250) — matrix operand order repaired and checked by reading, not compiled, not run on hardware
//
#include <hip/hip_runtime.h>
#include <math.h>

typedef __attribute__((ext_vector_type(16))) _Float16 v16h;
typedef __attribute__((ext_vector_type(8)))  float    v8f;

#define NS 32768
#define NF 128
#define MD 128
#define NE 16
#define NB 64
#define RESO 32
#define SRATE 22050.0f
#define PI_F 3.14159265358979323846f

// async global->LDS path (gfx1250): feature-detect so the build never breaks
#if defined(__has_builtin)
#if __has_builtin(__builtin_amdgcn_global_load_async_to_lds_b32) && \
    __has_builtin(__builtin_amdgcn_s_wait_asynccnt)
#define HAS_ASYNC_LDS 1
#endif
#endif
#ifndef HAS_ASYNC_LDS
#define HAS_ASYNC_LDS 0
#endif

#if HAS_ASYNC_LDS
typedef __attribute__((address_space(1))) int gint_t;   // global (AS1)
typedef __attribute__((address_space(3))) int lint_t;   // LDS (AS3)
#endif

// ---------------- workspace layout (bytes, 256-aligned) ----------------
constexpr size_t OFF_S16    = 0;                            // S^T: [f][k]  128x512 f16
constexpr size_t OFF_FBW16  = OFF_S16    + 512*128*2;
constexpr size_t OFF_POOLED = OFF_FBW16  + 128*512*2;
constexpr size_t OFF_SCAL   = OFF_POOLED + 128*128*4;
constexpr size_t OFF_BRED   = OFF_SCAL   + 256;             // B^T: [t][r]  128x192 f16
constexpr size_t OFF_ARED   = OFF_BRED   + 192*128*2;
constexpr size_t OFF_H      = OFF_ARED   + 128*192*2;
constexpr size_t OFF_BS     = OFF_H      + 128*128*4;       // B^T: [t][jc] 128x384 f16
constexpr size_t OFF_ADL    = OFF_BS     + 384*128*2;
constexpr size_t OFF_Y1     = OFF_ADL    + 128*384*2;
constexpr size_t OFF_Y1H    = OFF_Y1     + 128*128*4;       // B^T: [t][o]  128x128 f16
constexpr size_t OFF_PWH    = OFF_Y1H    + 128*128*2;
constexpr size_t OFF_Y2     = OFF_PWH    + 128*128*2;
constexpr size_t OFF_ENC16  = OFF_Y2     + 128*128*4;
constexpr size_t OFF_T16    = OFF_ENC16  + 16*128*2;
constexpr size_t OFF_TR16   = OFF_T16    + 16*128*2;
constexpr size_t OFF_PIMP   = OFF_TR16   + 16*128*2;
constexpr size_t OFF_IMP    = OFF_PIMP   + 16*128*4;
constexpr size_t OFF_TFL    = OFF_IMP    + 16*128*4;
constexpr size_t OFF_AMPS   = OFF_TFL    + 16*32*64*4;
constexpr size_t OFF_TTW16  = OFF_AMPS   + 16*64*128*4;     // already NxK
constexpr size_t OFF_TRW16  = OFF_TTW16  + 128*128*2;
constexpr size_t OFF_PLW16  = OFF_TRW16  + 128*128*2;
constexpr size_t OFF_ELW16  = OFF_PLW16  + 512*128*2;
constexpr size_t OFF_TLW16  = OFF_ELW16  + 512*128*2;
constexpr size_t OFF_ACT0   = OFF_TLW16  + 1024*128*2;
constexpr size_t OFF_ACT1   = OFF_ACT0   + (size_t)16*128*256*4;
constexpr size_t OFF_ENVS   = OFF_ACT1   + (size_t)16*128*256*4;
constexpr size_t OFF_TFM    = OFF_ENVS   + (size_t)16*NS*4;
constexpr size_t OFF_FINAL  = OFF_TFM    + (size_t)16*NS*4;

// ---------------- d_out layout (float elements) ----------------
constexpr int DO_OUT  = 0;       // (1,1,32768)
constexpr int DO_IDX  = 32768;   // (1,16) int32
constexpr int DO_ENC  = 32784;   // (1,16,128)
constexpr int DO_ENV  = 34832;   // (1,16,256)
constexpr int DO_TIME = 38928;   // (16,128)
constexpr int DO_TR   = 40976;   // (16,128)

static inline int gdiv(int n) { return (n + 255) / 256; }

// ============ f16 WMMA GEMM: C = act(A * BT^T + bias), 16x32 C tile / wave ============
// A: MxK row-major f16; BT: NxK row-major f16 (i.e. B stored transposed -> K-contiguous)
// biasMode: 0 none, 1 per-row(M), 2 per-col(N). act: 0 none, 1 leaky(0.2), 2 relu
__global__ void k_gemm_f16(const _Float16* __restrict__ A, const _Float16* __restrict__ BT,
                           float* __restrict__ C, const float* __restrict__ bias,
                           int M, int N, int K, int biasMode, int act)
{
  int wave = (int)((blockIdx.x * blockDim.x + threadIdx.x) >> 5);
  int tilesN = N >> 5;                       // 32-wide column tiles
  int tiles  = (M >> 4) * tilesN;
  if (wave >= tiles) return;
  int tm = wave / tilesN, tn = wave % tilesN;
  int lane = threadIdx.x & 31;
  int hh = lane >> 4;                        // K-half select
  int mm = lane & 15;
  int rowA = tm * 16 + mm;
  int col0 = tn * 32 + mm;
  int col1 = col0 + 16;
  const _Float16* arow = A  + (size_t)rowA * K;
  const _Float16* b0r  = BT + (size_t)col0 * K;
  const _Float16* b1r  = BT + (size_t)col1 * K;
  v8f acc0 = {}, acc1 = {};
  for (int k0 = 0; k0 < K; k0 += 32) {
    v16h af, bf0, bf1;
#pragma unroll
    for (int j = 0; j < 8; ++j) {   // A 16x32: VGPR j holds K = h*16 + 2j, 2j+1 (contiguous)
      int kb = k0 + hh * 16 + 2 * j;
      af[2 * j]     = arow[kb];
      af[2 * j + 1] = arow[kb + 1];
    }
#pragma unroll
    for (int j = 0; j < 8; ++j) {   // B 32x16: VGPR j holds K = h*16 + 2j, 2j+1 (contiguous)
      int kb = k0 + hh * 16 + 2 * j;
      bf0[2 * j]     = b0r[kb];
      bf0[2 * j + 1] = b0r[kb + 1];
      bf1[2 * j]     = b1r[kb];
      bf1[2 * j + 1] = b1r[kb + 1];
    }
    acc0 = __builtin_amdgcn_wmma_f32_16x16x32_f16(false, af, false, bf0,
                                                  (short)0, acc0, false, false);
    acc1 = __builtin_amdgcn_wmma_f32_16x16x32_f16(false, af, false, bf1,
                                                  (short)0, acc1, false, false);
  }
#pragma unroll
  for (int r = 0; r < 8; ++r) {     // C/D: VGPR r -> rows r (lanes0-15) / 8+r (lanes16-31)
    int row = tm * 16 + hh * 8 + r;
    float v0 = acc0[r], v1 = acc1[r];
    if (biasMode == 1) { float bb = bias[row]; v0 += bb; v1 += bb; }
    else if (biasMode == 2) { v0 += bias[col0]; v1 += bias[col1]; }
    if (act == 1) { v0 = (v0 > 0.f) ? v0 : 0.2f * v0; v1 = (v1 > 0.f) ? v1 : 0.2f * v1; }
    else if (act == 2) { v0 = fmaxf(v0, 0.f); v1 = fmaxf(v1, 0.f); }
    C[(size_t)row * N + col0] = v0;
    C[(size_t)row * N + col1] = v1;
  }
}

// ================= front-end: fused conv+avgpool sliding sums =================
// S[f,k] = (1/512) sum_{t in win_f} x_pad[t+k]; stored transposed: S16[f*512 + k]
__global__ void k_sliding(const float* __restrict__ x, _Float16* __restrict__ S16)
{
  __shared__ float xseg[1024];
  int f = blockIdx.x;
  int tlo = 256 * f - 256; if (tlo < 0) tlo = 0;
  int thi = 256 * f + 255; if (thi > NS - 1) thi = NS - 1;
  int winlen = thi - tlo + 1;
  int span = winlen + 511;
  for (int i = threadIdx.x; i < span; i += blockDim.x) {
    int xi = tlo + i - 256;  // x_pad has 256 left pad
    xseg[i] = (xi >= 0 && xi < NS) ? x[xi] : 0.0f;
  }
  __syncthreads();
  for (int k = threadIdx.x; k < 512; k += blockDim.x) {
    float s = 0.f;
    for (int i = 0; i < winlen; ++i) s += xseg[i + k];
    S16[f * 512 + k] = (_Float16)(s * (1.0f / 512.0f));
  }
}

__global__ void k_f32_to_f16(const float* __restrict__ s, _Float16* __restrict__ d, int n)
{
  int i = blockIdx.x * blockDim.x + threadIdx.x;
  if (i < n) d[i] = (_Float16)s[i];
}

// transpose-convert: src RxC f32 row-major -> dst CxR f16 row-major
__global__ void k_cvtT(const float* __restrict__ s, _Float16* __restrict__ d, int R, int Ccol)
{
  int i = blockIdx.x * blockDim.x + threadIdx.x;
  if (i >= R * Ccol) return;
  int r = i / Ccol, c = i % Ccol;
  d[(size_t)c * R + r] = (_Float16)s[i];
}

__global__ void k_sumsq(const float* __restrict__ a, int n, float* __restrict__ out)
{
  __shared__ float red[256];
  float s = 0.f;
  for (int i = threadIdx.x; i < n; i += blockDim.x) { float v = a[i]; s += v * v; }
  red[threadIdx.x] = s; __syncthreads();
  for (int st = 128; st > 0; st >>= 1) {
    if ((int)threadIdx.x < st) red[threadIdx.x] += red[threadIdx.x + st];
    __syncthreads();
  }
  if (threadIdx.x == 0) out[0] = red[0];
}

// B^T for reduce GEMM: BT[t, r], r<128: normalized pooled; 128..160: pos enc; rest 0
__global__ void k_build_bredT(const float* __restrict__ pooled, const float* __restrict__ scal,
                              _Float16* __restrict__ BT)
{
  int i = blockIdx.x * blockDim.x + threadIdx.x;
  if (i >= 128 * 192) return;
  int t = i / 192, r = i % 192;
  float val = 0.f;
  if (r < 128) {
    val = pooled[r * 128 + t] / (sqrtf(scal[0]) + 1e-8f);
  } else if (r < 161) {
    int pc = r - 128;
    float p = -1.0f + 2.0f * (float)t / 127.0f;
    if (pc == 0) val = p;
    else {
      int ib = (pc - 1) / 2;
      float ph = p * exp2f((float)ib) * PI_F;
      val = (pc & 1) ? sinf(ph) : cosf(ph);
    }
  }
  BT[i] = (_Float16)val;
}

// reduce_w (128x161) -> padded f16 A (128x192)
__global__ void k_padA(const float* __restrict__ w, _Float16* __restrict__ A)
{
  int i = blockIdx.x * blockDim.x + threadIdx.x;
  if (i >= 128 * 192) return;
  int o = i / 192, c = i % 192;
  A[i] = (_Float16)((c < 161) ? w[o * 161 + c] : 0.0f);
}

// A' for dilated conv-as-GEMM: A'[o, j*128+c] = dw[o,c,j]
__global__ void k_prep_dlA(const float* __restrict__ dw, _Float16* __restrict__ A)
{
  int i = blockIdx.x * blockDim.x + threadIdx.x;
  if (i >= 128 * 384) return;
  int o = i / 384, rest = i % 384;
  int j = rest / 128, c = rest % 128;
  A[i] = (_Float16)dw[(o * 128 + c) * 3 + j];
}

// B^T for dilated conv: Bs[t, j*128+c] = h[c, t + d*(j-1)] (zero pad)
__global__ void k_prep_BsT(const float* __restrict__ h, _Float16* __restrict__ Bs, int d)
{
  int i = blockIdx.x * blockDim.x + threadIdx.x;
  if (i >= 128 * 384) return;
  int t = i / 384, row = i % 384;
  int j = row / 128, c = row % 128;
  int tt = t + d * (j - 1);
  Bs[i] = (_Float16)((tt >= 0 && tt < 128) ? h[c * 128 + tt] : 0.0f);
}

__global__ void k_resleaky(float* __restrict__ h, const float* __restrict__ y, int n)
{
  int i = blockIdx.x * blockDim.x + threadIdx.x;
  if (i >= n) return;
  float v = h[i] + y[i];
  h[i] = (v > 0.f) ? v : 0.2f * v;
}

__global__ void k_scale(float* __restrict__ h, const float* __restrict__ scal, int n)
{
  int i = blockIdx.x * blockDim.x + threadIdx.x;
  if (i >= n) return;
  h[i] = h[i] / (sqrtf(scal[0]) + 1e-8f);
}

// norms over time, iterative top-16 (desc, first-index ties), encoded rows
__global__ void k_topk_encode(const float* __restrict__ h, float* __restrict__ dout,
                              _Float16* __restrict__ enc16)
{
  __shared__ float nrm[128];
  __shared__ float work[128];
  __shared__ int sidx[16];
  int q = threadIdx.x;  // time index for norms, channel index for encoded
  float s = 0.f;
  for (int c = 0; c < 128; ++c) { float v = h[c * 128 + q]; s += v * v; }
  float nv = sqrtf(s);
  nrm[q] = nv; work[q] = nv;
  __syncthreads();
  if (q == 0) {
    for (int j = 0; j < 16; ++j) {
      int bi = 0; float bv = work[0];
      for (int i = 1; i < 128; ++i) if (work[i] > bv) { bv = work[i]; bi = i; }
      sidx[j] = bi; work[bi] = -1e30f;
    }
  }
  __syncthreads();
  if (q < 16) ((int*)dout)[DO_IDX + q] = sidx[q];
  for (int j = 0; j < 16; ++j) {
    int ti = sidx[j];
    float v = h[q * 128 + ti] / (nrm[ti] + 1e-8f);
    dout[DO_ENC + j * 128 + q] = v;
    enc16[j * 128 + q] = (_Float16)v;
  }
}

// ConvTranspose1d k=4 s=2 p=1 (doubles T), leaky 0.2. w[(o*128+c)*4+j]
__global__ void k_convtr(const float* __restrict__ x, float* __restrict__ y,
                         const float* __restrict__ w, const float* __restrict__ b, int Tin)
{
  int Tout = Tin * 2;
  int gid = blockIdx.x * blockDim.x + threadIdx.x;
  if (gid >= NE * MD * Tout) return;
  int t = gid % Tout;
  int o = (gid / Tout) % MD;
  int e = gid / (Tout * MD);
  int m1, m2, j1, j2;
  if ((t & 1) == 0) { m1 = t / 2 - 1; j1 = 3; m2 = t / 2;       j2 = 1; }
  else              { m1 = (t - 1) / 2; j1 = 2; m2 = (t + 1) / 2; j2 = 0; }
  bool v1 = (m1 >= 0 && m1 < Tin);
  bool v2 = (m2 >= 0 && m2 < Tin);
  float acc = b[o];
  const float* xe = x + (size_t)e * MD * Tin;
  for (int c = 0; c < MD; ++c) {
    const float* wr = w + (o * MD + c) * 4;
    if (v1) acc += wr[j1] * xe[c * Tin + m1];
    if (v2) acc += wr[j2] * xe[c * Tin + m2];
  }
  y[gid] = (acc > 0.f) ? acc : 0.2f * acc;
}

// nearest x2 upsample + conv k=3 same pad, leaky 0.2. w[(o*128+c)*3+j]
__global__ void k_upconv(const float* __restrict__ x, float* __restrict__ y,
                         const float* __restrict__ w, const float* __restrict__ b, int Tin)
{
  int Tout = Tin * 2;
  int gid = blockIdx.x * blockDim.x + threadIdx.x;
  if (gid >= NE * MD * Tout) return;
  int t = gid % Tout;
  int o = (gid / Tout) % MD;
  int e = gid / (Tout * MD);
  float acc = b[o];
  const float* xe = x + (size_t)e * MD * Tin;
  for (int c = 0; c < MD; ++c) {
    const float* wr = w + (o * MD + c) * 3;
#pragma unroll
    for (int j = 0; j < 3; ++j) {
      int u = t + j - 1;
      if (u >= 0 && u < Tout) acc += wr[j] * xe[c * Tin + (u >> 1)];
    }
  }
  y[gid] = (acc > 0.f) ? acc : 0.2f * acc;
}

// conv k=3 same pad, MD->Oc channels. act: 0 none, 2 relu
__global__ void k_conv3out(const float* __restrict__ x, float* __restrict__ y,
                           const float* __restrict__ w, const float* __restrict__ b,
                           int Oc, int T, int act)
{
  int gid = blockIdx.x * blockDim.x + threadIdx.x;
  if (gid >= NE * Oc * T) return;
  int t = gid % T;
  int o = (gid / T) % Oc;
  int e = gid / (T * Oc);
  float acc = b[o];
  const float* xe = x + (size_t)e * MD * T;
  for (int c = 0; c < MD; ++c) {
    const float* wr = w + (o * MD + c) * 3;
#pragma unroll
    for (int j = 0; j < 3; ++j) {
      int u = t + j - 1;
      if (u >= 0 && u < T) acc += wr[j] * xe[c * T + u];
    }
  }
  if (act == 2) acc = fmaxf(acc, 0.f);
  y[gid] = acc;
}

__global__ void k_softmax128(const float* __restrict__ pin, float* __restrict__ pout)
{
  __shared__ float red[128];
  int e = blockIdx.x, t = threadIdx.x;
  float v = pin[e * 128 + t];
  red[t] = v; __syncthreads();
  for (int st = 64; st > 0; st >>= 1) { if (t < st) red[t] = fmaxf(red[t], red[t + st]); __syncthreads(); }
  float mx = red[0]; __syncthreads();
  float ex = expf(v - mx);
  red[t] = ex; __syncthreads();
  for (int st = 64; st > 0; st >>= 1) { if (t < st) red[t] += red[t + st]; __syncthreads(); }
  pout[e * 128 + t] = ex / red[0];
}

// envs[e,s] = lerp(env_e, s) * noise[s]
__global__ void k_interp(const float* __restrict__ dout, const float* __restrict__ noise,
                         float* __restrict__ envs)
{
  int gid = blockIdx.x * blockDim.x + threadIdx.x;
  if (gid >= NE * NS) return;
  int e = gid >> 15, s = gid & (NS - 1);
  float c = (s + 0.5f) * (256.0f / (float)NS) - 0.5f;
  c = fminf(fmaxf(c, 0.0f), 255.0f);
  int i0 = (int)floorf(c);
  int i1 = (i0 + 1 < 256) ? i0 + 1 : 255;
  float w = c - (float)i0;
  const float* env = dout + DO_ENV + e * 256;
  envs[gid] = (env[i0] * (1.0f - w) + env[i1] * w) * noise[s];
}

// hard straight-through select (forward == argmax one-hot) -> per-(e,band) decay table
__global__ void k_argmax_amp(const float* __restrict__ tfl, float* __restrict__ amp)
{
  int gid = blockIdx.x * blockDim.x + threadIdx.x;
  if (gid >= NE * NB) return;
  int e = gid / NB, b = gid % NB;
  int bi = 0; float bv = tfl[((size_t)e * RESO + 0) * NB + b];
  for (int r = 1; r < RESO; ++r) {
    float v = tfl[((size_t)e * RESO + r) * NB + b];
    if (v > bv) { bv = v; bi = r; }
  }
  float d = 0.5f + (float)bi * ((0.9999f - 0.5f) / 31.0f);
  float l2 = log2f(d);
  for (int fr = 0; fr < NF; ++fr)
    amp[((size_t)e * NB + b) * NF + fr] = exp2f((float)fr * l2);
}

// band-averaged selected bank signal: tfm[e,t] = (1/64) sum_b sin(w_b t) * amp[e,b,t>>8]
__global__ void k_tfmean(const float* __restrict__ amp, float* __restrict__ tfm)
{
  __shared__ float wf[NB];
  int t = blockIdx.x * 256 + threadIdx.x;
  if (threadIdx.x < NB) {
    float fb = 20.0f * powf(496.125f, (float)threadIdx.x / 63.0f); // geomspace(20, 9922.5, 64)
    wf[threadIdx.x] = 2.0f * PI_F * fb / SRATE;
  }
  __syncthreads();
  int fr = t >> 8;
  float acc[NE];
#pragma unroll
  for (int e = 0; e < NE; ++e) acc[e] = 0.f;
  for (int b = 0; b < NB; ++b) {
    float sv = sinf(wf[b] * (float)t);
    const float* ap = amp + (size_t)b * NF + fr;
#pragma unroll
    for (int e = 0; e < NE; ++e) acc[e] += sv * ap[(size_t)e * NB * NF];
  }
#pragma unroll
  for (int e = 0; e < NE; ++e) tfm[(size_t)e * NS + t] = acc[e] * (1.0f / 64.0f);
}

// causal linear convolution per event, LDS-tiled: final[e,s] = (1/256) sum_{tau<=s} envs[tau]*tfm[s-tau]
// envs chunk is staged via GLOBAL_LOAD_ASYNC_TO_LDS (ASYNCcnt) when available.
__global__ void k_bigconv(const float* __restrict__ envs, const float* __restrict__ tfm,
                          float* __restrict__ fin)
{
  __shared__ float esh[256];
  __shared__ float tsh[512];
  int e = blockIdx.y, sb = blockIdx.x;
  int tid = threadIdx.x;
  const float* ev = envs + (size_t)e * NS;
  const float* tf = tfm + (size_t)e * NS;
  int s0 = sb << 8;
  float acc = 0.f;
  for (int cb = 0; cb <= sb; ++cb) {
    int c0 = cb << 8;
    __syncthreads();
#if HAS_ASYNC_LDS
    __builtin_amdgcn_global_load_async_to_lds_b32(
        (gint_t*)const_cast<float*>(ev + c0 + tid),
        (lint_t*)&esh[tid], 0, 0);
#else
    esh[tid] = ev[c0 + tid];
#endif
    int base = s0 - c0 - 255;
    for (int i = tid; i < 512; i += 256) {
      int gi = base + i;
      tsh[i] = (gi >= 0 && gi < NS) ? tf[gi] : 0.f;   // zero-fill needs sync loads
    }
    if (cb < sb) __builtin_prefetch(ev + c0 + 256 + tid, 0, 1);  // global_prefetch_b8
#if HAS_ASYNC_LDS
    __builtin_amdgcn_s_wait_asynccnt(0);   // barrier does not drain ASYNCcnt
#endif
    __syncthreads();
#pragma unroll 8
    for (int j = 0; j < 256; ++j)
      acc = fmaf(esh[j], tsh[tid + 255 - j], acc);
  }
  fin[(size_t)e * NS + s0 + tid] = acc * (1.0f / 256.0f);
}

// sparse impulse comb (+ second fft_convolve scale) and sum over events
__global__ void k_output(const float* __restrict__ fin, const float* __restrict__ imp,
                         float* __restrict__ out)
{
  int t = blockIdx.x * blockDim.x + threadIdx.x;
  if (t >= NS) return;
  float acc = 0.f;
  int kmax = t >> 8; if (kmax > 127) kmax = 127;
  for (int e = 0; e < NE; ++e) {
    const float* fe = fin + (size_t)e * NS;
    const float* ie = imp + e * 128;
    for (int k = 0; k <= kmax; ++k) acc += ie[k] * fe[t - (k << 8)];
  }
  out[t] = acc * (1.0f / 256.0f);
}

// ============================== host driver ==============================
static void launch_gemm(hipStream_t s, const _Float16* A, const _Float16* BT, float* C,
                        const float* bias, int M, int N, int K, int bm, int act)
{
  int tiles = (M / 16) * (N / 32);
  int blocks = (tiles + 7) / 8;
  k_gemm_f16<<<blocks, 256, 0, s>>>(A, BT, C, bias, M, N, K, bm, act);
}

extern "C" void kernel_launch(void* const* d_in, const int* in_sizes, int n_in,
                              void* d_out, int out_size, void* d_ws, size_t ws_size,
                              hipStream_t stream)
{
  (void)in_sizes; (void)n_in; (void)out_size; (void)ws_size;
  const float* x         = (const float*)d_in[0];
  const float* noise     = (const float*)d_in[1];
  const float* fb_w      = (const float*)d_in[2];
  const float* reduce_w  = (const float*)d_in[3];
  const float* reduce_b  = (const float*)d_in[4];
  const float* dl_dw     = (const float*)d_in[5];
  const float* dl_db     = (const float*)d_in[6];
  const float* dl_pw     = (const float*)d_in[7];
  const float* dl_pb     = (const float*)d_in[8];
  const float* tt_w      = (const float*)d_in[9];
  const float* tt_b      = (const float*)d_in[10];
  const float* tr_w      = (const float*)d_in[11];
  const float* tr_b      = (const float*)d_in[12];
  const float* pos_lin_w = (const float*)d_in[13];
  const float* pos_lin_b = (const float*)d_in[14];
  const float* pos_up_w  = (const float*)d_in[15];
  const float* pos_up_b  = (const float*)d_in[16];
  const float* pos_out_w = (const float*)d_in[17];
  const float* pos_out_b = (const float*)d_in[18];
  const float* env_lin_w = (const float*)d_in[19];
  const float* env_lin_b = (const float*)d_in[20];
  const float* env_up_w  = (const float*)d_in[21];
  const float* env_up_b  = (const float*)d_in[22];
  const float* env_out_w = (const float*)d_in[23];
  const float* env_out_b = (const float*)d_in[24];
  const float* tfu_lin_w = (const float*)d_in[25];
  const float* tfu_lin_b = (const float*)d_in[26];
  const float* tfu_up_w  = (const float*)d_in[27];
  const float* tfu_up_b  = (const float*)d_in[28];
  const float* tfu_out_w = (const float*)d_in[29];
  const float* tfu_out_b = (const float*)d_in[30];

  char* W = (char*)d_ws;
  _Float16* S16   = (_Float16*)(W + OFF_S16);
  _Float16* FBW16 = (_Float16*)(W + OFF_FBW16);
  float*    POOL  = (float*)(W + OFF_POOLED);
  float*    SCAL  = (float*)(W + OFF_SCAL);
  _Float16* BRED  = (_Float16*)(W + OFF_BRED);
  _Float16* ARED  = (_Float16*)(W + OFF_ARED);
  float*    H     = (float*)(W + OFF_H);
  _Float16* BS    = (_Float16*)(W + OFF_BS);
  _Float16* ADL   = (_Float16*)(W + OFF_ADL);
  float*    Y1    = (float*)(W + OFF_Y1);
  _Float16* Y1H   = (_Float16*)(W + OFF_Y1H);
  _Float16* PWH   = (_Float16*)(W + OFF_PWH);
  float*    Y2    = (float*)(W + OFF_Y2);
  _Float16* ENC16 = (_Float16*)(W + OFF_ENC16);
  _Float16* T16   = (_Float16*)(W + OFF_T16);
  _Float16* TR16  = (_Float16*)(W + OFF_TR16);
  float*    PIMP  = (float*)(W + OFF_PIMP);
  float*    IMP   = (float*)(W + OFF_IMP);
  float*    TFL   = (float*)(W + OFF_TFL);
  float*    AMPS  = (float*)(W + OFF_AMPS);
  _Float16* TTW16 = (_Float16*)(W + OFF_TTW16);
  _Float16* TRW16 = (_Float16*)(W + OFF_TRW16);
  _Float16* PLW16 = (_Float16*)(W + OFF_PLW16);
  _Float16* ELW16 = (_Float16*)(W + OFF_ELW16);
  _Float16* TLW16 = (_Float16*)(W + OFF_TLW16);
  float*    ACT0  = (float*)(W + OFF_ACT0);
  float*    ACT1  = (float*)(W + OFF_ACT1);
  float*    ENVS  = (float*)(W + OFF_ENVS);
  float*    TFM   = (float*)(W + OFF_TFM);
  float*    FINAL = (float*)(W + OFF_FINAL);
  float*    dof   = (float*)d_out;

  // -------- weight f16 conversions (weights are naturally NxK for the BT layout) --------
  k_f32_to_f16<<<gdiv(128*512), 256, 0, stream>>>(fb_w, FBW16, 128*512);
  k_f32_to_f16<<<gdiv(128*128), 256, 0, stream>>>(tt_w, TTW16, 128*128);
  k_f32_to_f16<<<gdiv(128*128), 256, 0, stream>>>(tr_w, TRW16, 128*128);
  k_f32_to_f16<<<gdiv(512*128), 256, 0, stream>>>(pos_lin_w, PLW16, 512*128);
  k_f32_to_f16<<<gdiv(512*128), 256, 0, stream>>>(env_lin_w, ELW16, 512*128);
  k_f32_to_f16<<<gdiv(1024*128), 256, 0, stream>>>(tfu_lin_w, TLW16, 1024*128);

  // -------- front end: conv+pool fused as sliding-sum GEMM --------
  k_sliding<<<128, 256, 0, stream>>>(x, S16);
  launch_gemm(stream, FBW16, S16, POOL, nullptr, 128, 128, 512, 0, 0);
  k_sumsq<<<1, 256, 0, stream>>>(POOL, 128*128, SCAL);
  k_build_bredT<<<gdiv(128*192), 256, 0, stream>>>(POOL, SCAL, BRED);
  k_padA<<<gdiv(128*192), 256, 0, stream>>>(reduce_w, ARED);
  launch_gemm(stream, ARED, BRED, H, reduce_b, 128, 128, 192, 1, 0);

  // -------- dilated residual stack (conv-as-GEMM) --------
  static const int DIL[5] = {1, 3, 9, 27, 1};
  for (int L = 0; L < 5; ++L) {
    k_prep_dlA<<<gdiv(128*384), 256, 0, stream>>>(dl_dw + (size_t)L*128*128*3, ADL);
    k_prep_BsT<<<gdiv(128*384), 256, 0, stream>>>(H, BS, DIL[L]);
    launch_gemm(stream, ADL, BS, Y1, dl_db + L*128, 128, 128, 384, 1, 0);
    k_cvtT<<<gdiv(128*128), 256, 0, stream>>>(Y1, Y1H, 128, 128);
    k_f32_to_f16<<<gdiv(128*128), 256, 0, stream>>>(dl_pw + (size_t)L*128*128, PWH, 128*128);
    launch_gemm(stream, PWH, Y1H, Y2, dl_pb + L*128, 128, 128, 128, 1, 0);
    k_resleaky<<<gdiv(128*128), 256, 0, stream>>>(H, Y2, 128*128);
  }

  // -------- norm + top-k event selection --------
  k_sumsq<<<1, 256, 0, stream>>>(H, 128*128, SCAL);
  k_scale<<<gdiv(128*128), 256, 0, stream>>>(H, SCAL, 128*128);
  k_topk_encode<<<1, 128, 0, stream>>>(H, dof, ENC16);

  // -------- time / transfer projections (WMMA) --------
  launch_gemm(stream, ENC16, TTW16, dof + DO_TIME, tt_b, 16, 128, 128, 2, 0);
  launch_gemm(stream, ENC16, TRW16, dof + DO_TR,   tr_b, 16, 128, 128, 2, 0);
  k_f32_to_f16<<<gdiv(16*128), 256, 0, stream>>>(dof + DO_TIME, T16, 16*128);
  k_f32_to_f16<<<gdiv(16*128), 256, 0, stream>>>(dof + DO_TR, TR16, 16*128);

  // -------- impulse placement path --------
  launch_gemm(stream, T16, PLW16, ACT0, pos_lin_b, 16, 512, 128, 2, 0); // (16,128,4)
  {
    float* a = ACT0; float* b = ACT1; int T = 4;
    for (int L = 0; L < 5; ++L) {
      k_convtr<<<gdiv(NE*MD*2*T), 256, 0, stream>>>(a, b, pos_up_w + (size_t)L*128*128*4,
                                                    pos_up_b + L*128, T);
      float* tmp = a; a = b; b = tmp; T *= 2;
    } // a == result, T=128
    k_conv3out<<<gdiv(NE*128), 256, 0, stream>>>(a, PIMP, pos_out_w, pos_out_b, 1, 128, 0);
    k_softmax128<<<16, 128, 0, stream>>>(PIMP, IMP);
  }

  // -------- envelope path --------
  launch_gemm(stream, TR16, ELW16, ACT0, env_lin_b, 16, 512, 128, 2, 0); // (16,128,4)
  {
    float* a = ACT0; float* b = ACT1; int T = 4;
    for (int L = 0; L < 6; ++L) {
      k_upconv<<<gdiv(NE*MD*2*T), 256, 0, stream>>>(a, b, env_up_w + (size_t)L*128*128*3,
                                                    env_up_b + L*128, T);
      float* tmp = a; a = b; b = tmp; T *= 2;
    } // a == result, T=256
    k_conv3out<<<gdiv(NE*256), 256, 0, stream>>>(a, dof + DO_ENV, env_out_w, env_out_b, 1, 256, 2);
  }
  k_interp<<<gdiv(NE*NS), 256, 0, stream>>>(dof, noise, ENVS);

  // -------- transfer-function path --------
  launch_gemm(stream, TR16, TLW16, ACT0, tfu_lin_b, 16, 1024, 128, 2, 0); // (16,128,8)
  {
    float* a = ACT0; float* b = ACT1; int T = 8;
    for (int L = 0; L < 3; ++L) {
      k_upconv<<<gdiv(NE*MD*2*T), 256, 0, stream>>>(a, b, tfu_up_w + (size_t)L*128*128*3,
                                                    tfu_up_b + L*128, T);
      float* tmp = a; a = b; b = tmp; T *= 2;
    } // a == result, T=64
    k_conv3out<<<gdiv(NE*32*64), 256, 0, stream>>>(a, TFL, tfu_out_w, tfu_out_b, 32, 64, 0);
  }
  k_argmax_amp<<<gdiv(NE*NB), 256, 0, stream>>>(TFL, AMPS);
  k_tfmean<<<NS/256, 256, 0, stream>>>(AMPS, TFM);

  // -------- heavy convolutions + sparse comb --------
  k_bigconv<<<dim3(NS/256, NE), 256, 0, stream>>>(ENVS, TFM, FINAL);
  k_output<<<gdiv(NS), 256, 0, stream>>>(FINAL, IMP, dof + DO_OUT);
}
